// QEVCModel_824633721483
// MI455X (gfx1250) — matrix operand order repaired and checked by reading, not compiled, not run on hardware
//
#include <hip/hip_runtime.h>
#include <hip/hip_bf16.h>

// ---------------------------------------------------------------------------
// Problem constants
// ---------------------------------------------------------------------------
#define NQ    7
#define NL    6
#define NPCA  32
#define NCLS  3129
#define NCLSP 3136          // NCLS padded to multiple of 16 (and 64)
#define DIM   128           // 2^NQ
#define BATCH 8192

typedef _Float16 v16h __attribute__((ext_vector_type(16)));
typedef _Float16 v8h  __attribute__((ext_vector_type(8)));
typedef float    v8f  __attribute__((ext_vector_type(8)));

// ---------------------------------------------------------------------------
// Kernel 1: quantum circuit simulator.
// 256 threads/block = 2 samples/block, 128 threads (one complex amplitude)
// per sample. Gates via LDS exchanges; ring-CNOT layer composed into one
// permutation gather. Writes q_features (B x 7) into qf_out.
// ---------------------------------------------------------------------------
__global__ __launch_bounds__(256)
void quantum_kernel(const float* __restrict__ x,
                    const float* __restrict__ fscale,
                    const float* __restrict__ qw,   // (NL,NQ,2)
                    float* __restrict__ qf_out)     // (B,7)
{
    __shared__ float s_re[256];
    __shared__ float s_im[256];
    __shared__ float s_feat[2 * NL * NQ];

    const int tid = threadIdx.x;
    const int ls  = tid >> 7;          // local sample 0/1
    const int t   = tid & 127;         // amplitude index
    const int s   = blockIdx.x * 2 + ls;

    // precompute feats = atan(x*scale) padded to 42 per sample
    if (t < NL * NQ) {
        float f = 0.f;
        if (t < NPCA) f = atanf(x[s * NPCA + t] * fscale[t]);
        s_feat[ls * (NL * NQ) + t] = f;
    }
    __syncthreads();

    float re = (t == 0) ? 1.f : 0.f;
    float im = 0.f;

    for (int layer = 0; layer < NL; ++layer) {
        for (int q = 0; q < NQ; ++q) {
            const int bitpos = NQ - 1 - q;
            const int mybit  = (t >> bitpos) & 1;
            const int ppart  = (tid & 128) | (t ^ (1 << bitpos));

            // ---- RY(theta1) : butterfly via LDS ----
            s_re[tid] = re;
            s_im[tid] = im;
            __syncthreads();
            const float pr = s_re[ppart];
            const float pi = s_im[ppart];
            __syncthreads();

            const float th1 = s_feat[ls * (NL * NQ) + layer * NQ + q]
                            + qw[(layer * NQ + q) * 2 + 0];
            const float c = cosf(0.5f * th1);
            const float sn = sinf(0.5f * th1);
            float nre, nim;
            if (mybit == 0) {         // new0 = c*a0 - s*a1   (a0 = me)
                nre = c * re - sn * pr;
                nim = c * im - sn * pi;
            } else {                  // new1 = s*a0 + c*a1   (a1 = me)
                nre = sn * pr + c * re;
                nim = sn * pi + c * im;
            }
            re = nre; im = nim;

            // ---- RZ(theta2) : local phase ----
            const float th2 = qw[(layer * NQ + q) * 2 + 1];
            const float phi = (mybit == 0) ? -0.5f * th2 : 0.5f * th2;
            const float ca = cosf(phi), sa = sinf(phi);
            nre = re * ca - im * sa;
            nim = re * sa + im * ca;
            re = nre; im = nim;
        }

        // ---- composed ring-CNOT permutation: new[j] = old[P0(P1(..P6(j)))] ----
        s_re[tid] = re;
        s_im[tid] = im;
        __syncthreads();
        int src = t;
        for (int q = NQ - 1; q >= 0; --q) {
            const int cb = (src >> (NQ - 1 - q)) & 1;
            src ^= cb << (NQ - 1 - ((q + 1) % NQ));
        }
        const int psrc = (tid & 128) | src;
        re = s_re[psrc];
        im = s_im[psrc];
        __syncthreads();
    }

    // ---- probabilities and <Z_k> readout ----
    s_re[tid] = re * re + im * im;
    __syncthreads();
    if (t < NQ) {
        float acc = 0.f;
        const int base = ls * 128;
        #pragma unroll 1
        for (int k = 0; k < DIM; ++k) {
            const float sign = ((k >> (NQ - 1 - t)) & 1) ? -1.f : 1.f;
            acc += s_re[base + k] * sign;
        }
        qf_out[s * NQ + t] = acc;
    }
}

// ---------------------------------------------------------------------------
// Kernel 2: fp32 -> fp16 conversion (for activations)
// ---------------------------------------------------------------------------
__global__ void cvt_f16_kernel(const float* __restrict__ in,
                               _Float16* __restrict__ out, int n)
{
    int i = blockIdx.x * blockDim.x + threadIdx.x;
    if (i < n) out[i] = (_Float16)in[i];
}

// ---------------------------------------------------------------------------
// Kernel 3: weight transpose+convert: (K x N) f32 row-major -> (Npad x K) f16,
// zero-filling padded rows.
// ---------------------------------------------------------------------------
__global__ void transcvt_kernel(const float* __restrict__ in,
                                _Float16* __restrict__ out,
                                int K, int N, int Npad)
{
    int i = blockIdx.x * blockDim.x + threadIdx.x;
    if (i >= Npad * K) return;
    int n = i / K, k = i - n * K;
    out[i] = (n < N) ? (_Float16)in[k * N + n] : (_Float16)0.f;
}

// ---------------------------------------------------------------------------
// Kernel 4: tiny K=7 quantum head layer 1: hq = relu(qf @ qh_w1 + b1) -> f16
// ---------------------------------------------------------------------------
__global__ void qhead1_kernel(const float* __restrict__ qf,   // (B,7)
                              const float* __restrict__ w1,   // (7,64)
                              const float* __restrict__ b1,   // (64)
                              _Float16* __restrict__ hq)      // (B,64) f16
{
    int i = blockIdx.x * blockDim.x + threadIdx.x;   // B*64 threads
    int m = i >> 6, j = i & 63;
    float acc = b1[j];
    #pragma unroll
    for (int k = 0; k < NQ; ++k)
        acc += qf[m * NQ + k] * w1[k * 64 + j];
    hq[i] = (_Float16)fmaxf(acc, 0.f);
}

// ---------------------------------------------------------------------------
// Blocked WMMA GEMM: each wave owns a 64x64 macro-tile (4x4 WMMA tiles).
// D = A(MxK f16 row-major) @ Bt^T (Bt is Npad x K f16 row-major) + bias.
// Per 32-wide K chunk: 4 A-fragments + 4 B-fragments -> 16 WMMAs, i.e. 4x
// register-level fragment reuse vs one-tile-per-wave (keeps L2 fragment
// traffic under the HBM output-store floor).
// RELU_F16=0 : f32 store to outF with n<N guard (logits)
// RELU_F16=1 : relu + f16 store to outH (hidden layers, N==Npad)
// Requires M % 64 == 0 and Npad % 64 == 0.
// ---------------------------------------------------------------------------
template <int RELU_F16>
__global__ __launch_bounds__(256)
void gemm_wmma_kernel(const _Float16* __restrict__ A,
                      const _Float16* __restrict__ Bt,
                      const float* __restrict__ bias,
                      float* __restrict__ outF,
                      _Float16* __restrict__ outH,
                      int N, int Npad, int K)
{
    const int lane    = threadIdx.x & 31;
    const int waveId  = blockIdx.x * 8 + (threadIdx.x >> 5);
    const int tilesN  = Npad >> 6;              // 64-wide macro-tiles in N
    const int tm      = waveId / tilesN;
    const int tn      = waveId - tm * tilesN;
    const int m0      = tm << 6;
    const int n0      = tn << 6;

    const int lane15 = lane & 15;
    const int half   = lane >> 4;

    // per-lane fragment row pointers (A: K = half*8+{0..7} then +16;
    //                                 B: K = half*8+{0..7} then +16)
    const _Float16* arow[4];
    const _Float16* brow[4];
    #pragma unroll
    for (int i = 0; i < 4; ++i)
        arow[i] = A + (long)(m0 + i * 16 + lane15) * K + half * 8;
    #pragma unroll
    for (int j = 0; j < 4; ++j)
        brow[j] = Bt + (long)(n0 + j * 16 + lane15) * K + half * 8;

    v8f acc[4][4];
    #pragma unroll
    for (int i = 0; i < 4; ++i)
        #pragma unroll
        for (int j = 0; j < 4; ++j)
            acc[i][j] = (v8f){};

    for (int kc = 0; kc < K; kc += 32) {
        v16h af[4], bf[4];
        #pragma unroll
        for (int i = 0; i < 4; ++i) {
            v8h lo = *(const v8h*)(arow[i] + kc);
            v8h hi = *(const v8h*)(arow[i] + kc + 16);
            af[i] = __builtin_shufflevector(lo, hi,
                        0,1,2,3,4,5,6,7, 8,9,10,11,12,13,14,15);
        }
        #pragma unroll
        for (int j = 0; j < 4; ++j) {
            v8h lo = *(const v8h*)(brow[j] + kc);
            v8h hi = *(const v8h*)(brow[j] + kc + 16);
            bf[j] = __builtin_shufflevector(lo, hi,
                        0,1,2,3,4,5,6,7, 8,9,10,11,12,13,14,15);
        }
        #pragma unroll
        for (int i = 0; i < 4; ++i)
            #pragma unroll
            for (int j = 0; j < 4; ++j)
                acc[i][j] = __builtin_amdgcn_wmma_f32_16x16x32_f16(
                                false, af[i], false, bf[j],
                                (short)0, acc[i][j], false, false);
    }

    // D layout per 16x16 tile: VGPR r -> row (+r + half*8), col lane15
    #pragma unroll
    for (int j = 0; j < 4; ++j) {
        const int n = n0 + j * 16 + lane15;
        if (RELU_F16) {
            const float bv = bias[n];
            #pragma unroll
            for (int i = 0; i < 4; ++i) {
                const int mrow = m0 + i * 16 + half * 8;
                #pragma unroll
                for (int r = 0; r < 8; ++r) {
                    float v = acc[i][j][r] + bv;
                    outH[(long)(mrow + r) * N + n] = (_Float16)fmaxf(v, 0.f);
                }
            }
        } else {
            if (n < N) {
                const float bv = bias[n];
                #pragma unroll
                for (int i = 0; i < 4; ++i) {
                    const int mrow = m0 + i * 16 + half * 8;
                    #pragma unroll
                    for (int r = 0; r < 8; ++r)
                        outF[(long)(mrow + r) * N + n] = acc[i][j][r] + bv;
                }
            }
        }
    }
}

// ---------------------------------------------------------------------------
// Launch
// ---------------------------------------------------------------------------
extern "C" void kernel_launch(void* const* d_in, const int* in_sizes, int n_in,
                              void* d_out, int out_size, void* d_ws, size_t ws_size,
                              hipStream_t stream)
{
    const float* x      = (const float*)d_in[0];
    const float* fscale = (const float*)d_in[1];
    const float* qw     = (const float*)d_in[2];
    const float* qh_w1  = (const float*)d_in[3];
    const float* qh_b1  = (const float*)d_in[4];
    const float* qh_w2  = (const float*)d_in[5];   // (64,3129)
    const float* qh_b2  = (const float*)d_in[6];
    const float* ch_w1  = (const float*)d_in[7];   // (32,256)
    const float* ch_b1  = (const float*)d_in[8];
    const float* ch_w2  = (const float*)d_in[9];   // (256,128)
    const float* ch_b2  = (const float*)d_in[10];
    const float* ch_w3  = (const float*)d_in[11];  // (128,3129)
    const float* ch_b3  = (const float*)d_in[12];

    float* q_logits = (float*)d_out;
    float* c_logits = q_logits + (size_t)BATCH * NCLS;
    float* qf       = c_logits + (size_t)BATCH * NCLS;   // (B,7)

    // workspace layout (f16 buffers), 256B aligned offsets
    char* ws = (char*)d_ws;
    size_t off = 0;
    auto carve = [&](size_t bytes) {
        char* p = ws + off;
        off = (off + bytes + 255) & ~(size_t)255;
        return p;
    };
    _Float16* x16   = (_Float16*)carve((size_t)BATCH * NPCA * 2);     // 512KB
    _Float16* w1t   = (_Float16*)carve((size_t)256   * 32   * 2);     // ch_w1^T
    _Float16* w2t   = (_Float16*)carve((size_t)128   * 256  * 2);     // ch_w2^T
    _Float16* w3t   = (_Float16*)carve((size_t)NCLSP * 128  * 2);     // ch_w3^T
    _Float16* qw2t  = (_Float16*)carve((size_t)NCLSP * 64   * 2);     // qh_w2^T
    _Float16* hq16  = (_Float16*)carve((size_t)BATCH * 64   * 2);
    _Float16* h1_16 = (_Float16*)carve((size_t)BATCH * 256  * 2);
    _Float16* h2_16 = (_Float16*)carve((size_t)BATCH * 128  * 2);
    (void)ws_size;

    // 1. quantum circuit -> qf
    quantum_kernel<<<BATCH / 2, 256, 0, stream>>>(x, fscale, qw, qf);

    // 2. conversions / transposes (independent of circuit)
    cvt_f16_kernel<<<(BATCH * NPCA + 255) / 256, 256, 0, stream>>>(x, x16, BATCH * NPCA);
    transcvt_kernel<<<(256 * 32 + 255) / 256, 256, 0, stream>>>(ch_w1, w1t, 32, 256, 256);
    transcvt_kernel<<<(128 * 256 + 255) / 256, 256, 0, stream>>>(ch_w2, w2t, 256, 128, 128);
    transcvt_kernel<<<(NCLSP * 128 + 255) / 256, 256, 0, stream>>>(ch_w3, w3t, 128, NCLS, NCLSP);
    transcvt_kernel<<<(NCLSP * 64 + 255) / 256, 256, 0, stream>>>(qh_w2, qw2t, 64, NCLS, NCLSP);

    // 3. quantum head layer1 (K=7, scalar) -> hq16
    qhead1_kernel<<<(BATCH * 64) / 256, 256, 0, stream>>>(qf, qh_w1, qh_b1, hq16);

    // 4. classical head, blocked WMMA chain (64x64 macro-tile per wave)
    // h1 = relu(x16 @ w1t^T + b1)   M=8192 N=256 K=32 : 128*4 = 512 waves
    gemm_wmma_kernel<1><<<(128 * 4) / 8, 256, 0, stream>>>(
        x16, w1t, ch_b1, nullptr, h1_16, 256, 256, 32);
    // h2 = relu(h1 @ w2t^T + b2)    M=8192 N=128 K=256 : 128*2 = 256 waves
    gemm_wmma_kernel<1><<<(128 * 2) / 8, 256, 0, stream>>>(
        h1_16, w2t, ch_b2, nullptr, h2_16, 128, 128, 256);
    // c_logits = h2 @ w3t^T + b3    M=8192 N=3129(->3136) K=128 : 128*49 waves
    gemm_wmma_kernel<0><<<(128 * (NCLSP / 64)) / 8, 256, 0, stream>>>(
        h2_16, w3t, ch_b3, c_logits, nullptr, NCLS, NCLSP, 128);

    // 5. q_logits = hq @ qw2t^T + b2   M=8192 N=3129(->3136) K=64
    gemm_wmma_kernel<0><<<(128 * (NCLSP / 64)) / 8, 256, 0, stream>>>(
        hq16, qw2t, qh_b2, q_logits, nullptr, NCLS, NCLSP, 64);
}
